// SpatialOCR_Module_38611755991121
// MI455X (gfx1250) — matrix operand order repaired and checked by reading, not compiled, not run on hardware
//
#include <hip/hip_runtime.h>
#include <math.h>

typedef __attribute__((ext_vector_type(16))) _Float16 v16h;
typedef __attribute__((ext_vector_type(8)))  float    v8f;

union HFrag {
    v16h h;
    uint4 q[2];
    _Float16 e[16];
};

// 16 contiguous halfs (B-fragment segment): e[i] = K = i (+16*hh applied by caller)
__device__ __forceinline__ HFrag load_frag_h16(const _Float16* p) {
    HFrag f;
    f.q[0] = *(const uint4*)(p);
    f.q[1] = *(const uint4*)(p + 8);
    return f;
}

// A-fragment (16x32 f16) from a row of contiguous halfs:
//   e[0..15] <- K = 16*hh .. 16*hh+15
__device__ __forceinline__ HFrag load_afrag_h(const _Float16* row, int hh) {
    HFrag f;
    f.q[0] = *(const uint4*)(row + 16 * hh);
    f.q[1] = *(const uint4*)(row + 16 * hh + 8);
    return f;
}

// 8 f32 at p0 -> e[0..7], 8 f32 at p1 -> e[8..15] (converted to f16)
__device__ __forceinline__ HFrag load_frag_f32(const float* p0, const float* p1) {
    HFrag f;
    float4 a = *(const float4*)(p0);
    float4 b = *(const float4*)(p0 + 4);
    float4 c = *(const float4*)(p1);
    float4 d = *(const float4*)(p1 + 4);
    f.e[0]  = (_Float16)a.x; f.e[1]  = (_Float16)a.y;
    f.e[2]  = (_Float16)a.z; f.e[3]  = (_Float16)a.w;
    f.e[4]  = (_Float16)b.x; f.e[5]  = (_Float16)b.y;
    f.e[6]  = (_Float16)b.z; f.e[7]  = (_Float16)b.w;
    f.e[8]  = (_Float16)c.x; f.e[9]  = (_Float16)c.y;
    f.e[10] = (_Float16)c.z; f.e[11] = (_Float16)c.w;
    f.e[12] = (_Float16)d.x; f.e[13] = (_Float16)d.y;
    f.e[14] = (_Float16)d.z; f.e[15] = (_Float16)d.w;
    return f;
}

#define WMMA_F16(A, B, C) \
    __builtin_amdgcn_wmma_f32_16x16x32_f16(false, (A), false, (B), (short)0, (C), false, false)

// CDNA5 async copy global -> LDS (no VGPR round trip, tracked by ASYNCcnt)
__device__ __forceinline__ void async_b128(unsigned lds_addr, const void* gptr) {
    asm volatile("global_load_async_to_lds_b128 %0, %1, off"
                 :: "v"(lds_addr), "v"(gptr)
                 : "memory");
}
__device__ __forceinline__ void wait_async0() {
    asm volatile("s_wait_asynccnt 0x0" ::: "memory");
}

#define CIN 512
#define CKD 256

// out1[b][pix][CKD] = (w1 @ x[b,:,pix] + b1) * scale1   (f16)
// out2t[b][ck][pix] = (w2 @ x[b,:,pix] + b2)            (f16, transposed) -- optional
__global__ __launch_bounds__(256) void proj_kernel(
    const float* __restrict__ x,   // [B][CIN][Npix]
    const float* __restrict__ w1, const float* __restrict__ b1,
    const float* __restrict__ w2, const float* __restrict__ b2,
    _Float16* __restrict__ o1,     // [B][Npix][CKD]
    _Float16* __restrict__ o2t,    // [B][CKD][Npix] or nullptr
    int Npix, float scale1)
{
    __shared__ _Float16 xs[32][32]; // [pix_local][c_local]
    const int tid  = threadIdx.x;
    const int lane = tid & 31;
    const int wv   = tid >> 5;
    const int hh   = lane >> 4;
    const int lc   = lane & 15;
    const int m0   = blockIdx.x * 32;
    const int ckb  = blockIdx.y * 64;
    const int b    = blockIdx.z;
    const int mt   = wv & 1;       // pixel sub-tile (16)
    const int ckt  = wv >> 1;      // ck sub-tile (16)
    const int ck0  = ckb + ckt * 16;
    const int mr0  = m0 + mt * 16;
    const bool dual = (o2t != nullptr);

    v8f acc1 = {};
    v8f acc2 = {};

    const int cl = tid >> 3;        // 0..31 channel
    const int ml = (tid & 7) * 4;   // 0..28 pixel

    for (int c0 = 0; c0 < CIN; c0 += 32) {
        // cooperative stage: [32 c][32 pix] f32 -> LDS [pix][c] f16 (transpose)
        float4 v = *(const float4*)(x + ((size_t)b * CIN + c0 + cl) * Npix + m0 + ml);
        xs[ml + 0][cl] = (_Float16)v.x;
        xs[ml + 1][cl] = (_Float16)v.y;
        xs[ml + 2][cl] = (_Float16)v.z;
        xs[ml + 3][cl] = (_Float16)v.w;
        __syncthreads();

        HFrag a = load_afrag_h(&xs[mt * 16 + lc][0], hh);

        const float* wp1 = w1 + (size_t)(ck0 + lc) * CIN + c0 + 16 * hh;
        HFrag bw1 = load_frag_f32(wp1, wp1 + 8);
        acc1 = WMMA_F16(a.h, bw1.h, acc1);
        if (dual) {
            const float* wp2 = w2 + (size_t)(ck0 + lc) * CIN + c0 + 16 * hh;
            HFrag bw2 = load_frag_f32(wp2, wp2 + 8);
            acc2 = WMMA_F16(a.h, bw2.h, acc2);
        }
        __syncthreads();
    }

    const int col = ck0 + lc;
    const float bias1 = b1[col];
#pragma unroll
    for (int r = 0; r < 8; ++r) {
        const int m = mr0 + r + 8 * hh;
        o1[((size_t)b * Npix + m) * CKD + col] = (_Float16)((acc1[r] + bias1) * scale1);
    }
    if (dual) {
        const float bias2 = b2[col];
#pragma unroll
        for (int r = 0; r < 8; ++r) {
            const int m = mr0 + r + 8 * hh;
            o2t[((size_t)b * CKD + col) * Npix + m] = (_Float16)(acc2[r] + bias2);
        }
    }
}

// Issue the async stage of one 32-wide M chunk of K and V into LDS.
// K chunk: rows [m0, m0+32) of Km[b] -> 16KB contiguous; thread copies 128B.
// V chunk: 256 rows x 64B (row stride 2048B in Vt[b]); thread copies 2 rows.
__device__ __forceinline__ void issue_chunk(const char* gK, const char* gV,
                                            unsigned ldsK, unsigned ldsV,
                                            int m0, int tid)
{
#pragma unroll
    for (int i = 0; i < 8; ++i)
        async_b128(ldsK + tid * 128 + i * 16,
                   gK + (size_t)m0 * 512 + tid * 128 + i * 16);
#pragma unroll
    for (int rr = 0; rr < 2; ++rr) {
        const int row = 2 * tid + rr;
#pragma unroll
        for (int i = 0; i < 4; ++i)
            async_b128(ldsV + row * 64 + i * 16,
                       gV + (size_t)row * 2048 + m0 * 2 + i * 16);
    }
}

// Flash attention: per wave, 16 query rows x full Ck=256; softmax in exp2 domain
// (log2e * Ck^-0.5 pre-folded into Q). K/V chunks double-buffered in LDS via
// global_load_async_to_lds_b128 (shared by the 4 waves of the block).
__global__ __launch_bounds__(128) void attn_kernel(
    const _Float16* __restrict__ Q,   // [B][4096][256]
    const _Float16* __restrict__ Km,  // [B][1024][256]
    const _Float16* __restrict__ Vt,  // [B][256][1024]
    _Float16* __restrict__ ctxo)      // [B][4096][256]
{
    __shared__ _Float16 Kbuf[2][32 * 256];   // [m_local][ck]
    __shared__ _Float16 Vbuf[2][256 * 32];   // [ck][m_local]
    __shared__ _Float16 Pbuf[4][16][32];

    const int tid  = threadIdx.x;
    const int lane = tid & 31;
    const int wv   = tid >> 5;
    const int hh   = lane >> 4;
    const int lc   = lane & 15;
    const int gw   = blockIdx.x * 4 + wv;
    const int b    = gw >> 8;          // 256 row-tiles per batch; 4 waves share b
    const int n0   = (gw & 255) * 16;

    const char* gK = (const char*)(Km + (size_t)b * 1024 * 256);
    const char* gV = (const char*)(Vt + (size_t)b * 256 * 1024);
    // low 32 bits of a flat __shared__ pointer == LDS address (ISA 10.2 aperture)
    const unsigned ldsK[2] = { (unsigned)(uintptr_t)&Kbuf[0][0],
                               (unsigned)(uintptr_t)&Kbuf[1][0] };
    const unsigned ldsV[2] = { (unsigned)(uintptr_t)&Vbuf[0][0],
                               (unsigned)(uintptr_t)&Vbuf[1][0] };

    // Q tile: 8 A-fragments over Ck (held in VGPRs for all 32 chunks)
    HFrag qa[8];
    const _Float16* qrow = Q + ((size_t)b * 4096 + n0 + lc) * 256;
#pragma unroll
    for (int kk = 0; kk < 8; ++kk)
        qa[kk] = load_afrag_h(qrow + kk * 32, hh);

    v8f O[16];
#pragma unroll
    for (int t = 0; t < 16; ++t) {
#pragma unroll
        for (int r = 0; r < 8; ++r) O[t][r] = 0.0f;
    }
    float mrow[8], lrow[8];
#pragma unroll
    for (int r = 0; r < 8; ++r) { mrow[r] = -1e30f; lrow[r] = 0.0f; }

    issue_chunk(gK, gV, ldsK[0], ldsV[0], 0, tid);

    for (int m0 = 0; m0 < 1024; m0 += 32) {
        const int buf = (m0 >> 5) & 1;
        wait_async0();          // my chunk's async copies landed in LDS
        __syncthreads();        // everyone's did
        if (m0 + 32 < 1024)     // prefetch next chunk into the other buffer
            issue_chunk(gK, gV, ldsK[buf ^ 1], ldsV[buf ^ 1], m0 + 32, tid);

        // S = Q x K_chunk  (two 16x16 tiles over 32 columns)
        v8f S0 = {}, S1 = {};
        const _Float16* kb0 = &Kbuf[buf][0] + (size_t)lc * 256 + 16 * hh;
        const _Float16* kb1 = kb0 + (size_t)16 * 256;
#pragma unroll
        for (int kk = 0; kk < 8; ++kk) {
            HFrag f0 = load_frag_h16(kb0 + kk * 32);
            S0 = WMMA_F16(qa[kk].h, f0.h, S0);
            HFrag f1 = load_frag_h16(kb1 + kk * 32);
            S1 = WMMA_F16(qa[kk].h, f1.h, S1);
        }

        float corr[8];
#pragma unroll
        for (int r = 0; r < 8; ++r) {
            // row max over this 32-wide chunk; xor masks <16 stay inside each
            // 16-lane half, which is exactly one row of the C/D tile layout.
            float t = fmaxf(S0[r], S1[r]);
            t = fmaxf(t, __shfl_xor(t, 1));
            t = fmaxf(t, __shfl_xor(t, 2));
            t = fmaxf(t, __shfl_xor(t, 4));
            t = fmaxf(t, __shfl_xor(t, 8));
            const float mn = fmaxf(mrow[r], t);
            corr[r] = exp2f(mrow[r] - mn);
            mrow[r] = mn;
            const float p0 = exp2f(S0[r] - mn);
            const float p1 = exp2f(S1[r] - mn);
            float s = p0 + p1;
            s += __shfl_xor(s, 1);
            s += __shfl_xor(s, 2);
            s += __shfl_xor(s, 4);
            s += __shfl_xor(s, 8);
            lrow[r] = lrow[r] * corr[r] + s;
            const int row = r + 8 * hh;
            Pbuf[wv][row][lc]      = (_Float16)p0;
            Pbuf[wv][row][lc + 16] = (_Float16)p1;
        }
#pragma unroll
        for (int t = 0; t < 16; ++t) {
#pragma unroll
            for (int r = 0; r < 8; ++r) O[t][r] *= corr[r];
        }

        // P relayout C/D -> A through per-wave LDS (same-wave DS ops are in order)
        HFrag pa = load_afrag_h(&Pbuf[wv][lc][0], hh);

        // O += P x V_chunk  (16 output tiles along Ck)
        const _Float16* vb = &Vbuf[buf][0] + (size_t)lc * 32 + 16 * hh;
#pragma unroll
        for (int t = 0; t < 16; ++t) {
            HFrag f = load_frag_h16(vb + (size_t)t * 16 * 32);
            O[t] = WMMA_F16(pa.h, f.h, O[t]);
        }
        __syncthreads();  // all waves done with buf before it is refilled
    }

    float rinv[8];
#pragma unroll
    for (int r = 0; r < 8; ++r) rinv[r] = 1.0f / lrow[r];

    _Float16* orow = ctxo + ((size_t)b * 4096 + n0) * 256;
#pragma unroll
    for (int t = 0; t < 16; ++t) {
#pragma unroll
        for (int r = 0; r < 8; ++r)
            orow[(size_t)(r + 8 * hh) * 256 + t * 16 + lc] = (_Float16)(O[t][r] * rinv[r]);
    }
}

// out = BN( wo @ ctx + bo ) with inference affine folded in; f32 output [B][512][4096]
__global__ __launch_bounds__(256) void outproj_kernel(
    const _Float16* __restrict__ ctxb, // [B][4096][256]
    const float* __restrict__ wo,      // [512][256]
    const float* __restrict__ bo,
    const float* __restrict__ gamma,
    const float* __restrict__ beta,
    const float* __restrict__ rmean,
    const float* __restrict__ rvar,
    float* __restrict__ out)           // [B][512][4096]
{
    const int lane = threadIdx.x & 31;
    const int wv   = threadIdx.x >> 5;
    const int hh   = lane >> 4;
    const int lc   = lane & 15;
    const int gw   = blockIdx.x * 8 + wv;
    const int b    = gw >> 11;                 // 2048 tiles per batch
    const int rem  = gw & 2047;
    const int co0  = (rem >> 6) * 16;          // 32 co-tiles
    const int n0   = (rem & 63) * 64;          // 64 n-spans of 64

    v8f acc[4];
#pragma unroll
    for (int t = 0; t < 4; ++t) {
#pragma unroll
        for (int r = 0; r < 8; ++r) acc[t][r] = 0.0f;
    }

    const float* wrow = wo + (size_t)(co0 + lc) * 256;
    const _Float16* cbase = ctxb + ((size_t)b * 4096 + n0 + lc) * 256 + 16 * hh;
#pragma unroll
    for (int kk = 0; kk < 8; ++kk) {
        const float* wp = wrow + kk * 32;
        HFrag a = load_frag_f32(wp + 16 * hh, wp + 16 * hh + 8);
#pragma unroll
        for (int t = 0; t < 4; ++t) {
            HFrag bf = load_frag_h16(cbase + (size_t)t * 16 * 256 + kk * 32);
            acc[t] = WMMA_F16(a.h, bf.h, acc[t]);
        }
    }

#pragma unroll
    for (int r = 0; r < 8; ++r) {
        const int co = co0 + r + 8 * hh;
        const float inv = gamma[co] / sqrtf(rvar[co] + 1e-5f);
        const float sh  = beta[co] + (bo[co] - rmean[co]) * inv;
        float* op = out + ((size_t)b * 512 + co) * 4096 + n0 + lc;
#pragma unroll
        for (int t = 0; t < 4; ++t)
            op[t * 16] = acc[t][r] * inv + sh;
    }
}

extern "C" void kernel_launch(void* const* d_in, const int* in_sizes, int n_in,
                              void* d_out, int out_size, void* d_ws, size_t ws_size,
                              hipStream_t stream) {
    (void)in_sizes; (void)n_in; (void)out_size; (void)ws_size;
    const float* features = (const float*)d_in[0];
    const float* context  = (const float*)d_in[1];
    const float* wq = (const float*)d_in[2];
    const float* bq = (const float*)d_in[3];
    const float* wk = (const float*)d_in[4];
    const float* bk = (const float*)d_in[5];
    const float* wv = (const float*)d_in[6];
    const float* bv = (const float*)d_in[7];
    const float* wo = (const float*)d_in[8];
    const float* bo = (const float*)d_in[9];
    const float* gamma = (const float*)d_in[10];
    const float* beta  = (const float*)d_in[11];
    const float* rmean = (const float*)d_in[12];
    const float* rvar  = (const float*)d_in[13];
    float* out = (float*)d_out;

    char* ws = (char*)d_ws;
    _Float16* Qb = (_Float16*)(ws);                                  // 16 MB
    _Float16* Kb = (_Float16*)(ws + (size_t)16 * 1024 * 1024);       //  4 MB
    _Float16* Vb = (_Float16*)(ws + (size_t)20 * 1024 * 1024);       //  4 MB
    _Float16* Cb = (_Float16*)(ws + (size_t)24 * 1024 * 1024);       // 16 MB

    // softmax scale (Ck^-0.5) and exp->exp2 conversion folded into Q
    const float qscale = 0.0625f * 1.44269504088896341f;

    // context -> K [B,M,Ck] and V^T [B,Ck,M]
    proj_kernel<<<dim3(32, 4, 8), 256, 0, stream>>>(
        context, wk, bk, wv, bv, Kb, Vb, 1024, 1.0f);
    // features -> Q [B,N,Ck] (pre-scaled)
    proj_kernel<<<dim3(128, 4, 8), 256, 0, stream>>>(
        features, wq, bq, nullptr, nullptr, Qb, nullptr, 4096, qscale);
    // flash attention -> ctx [B,N,Ck]
    attn_kernel<<<512, 128, 0, stream>>>(Qb, Kb, Vb, Cb);
    // output projection + BN
    outproj_kernel<<<2048, 256, 0, stream>>>(
        Cb, wo, bo, gamma, beta, rmean, rvar, out);
}
